// CorticalMap_18949395710658
// MI455X (gfx1250) — matrix operand order repaired and checked by reading, not compiled, not run on hardware
//
#include <hip/hip_runtime.h>
#include <hip/hip_bf16.h>
#include <cmath>

typedef __attribute__((ext_vector_type(16))) _Float16 v16h;
typedef __attribute__((ext_vector_type(8)))  float    v8f;

#define SHEET 128
#define L_UNITS (SHEET*SHEET)      // 16384
#define RF_K 15
#define RF_J (RF_K*RF_K)           // 225
#define SRE_KS 5
#define LRI_K 55
#define LRI_J (LRI_K*LRI_K)        // 3025
#define LAT_ITERS 10
#define ZPAD 27                    // LRI_K/2 * LAT_SPARSITY
#define ZDIM (SHEET + 2*ZPAD)      // 182 (row width)
#define ZROWS 184                  // 182 + 2 zeroed guard rows for padded K
#define KCHUNKS 95                 // 3025 padded to 3040 = 95 * 32
#define KPAD (KCHUNKS*32)          // 3040
#define NGROUPS (L_UNITS/16)       // 1024 groups of 16 column-adjacent units

__device__ __forceinline__ float wred(float v) {
    #pragma unroll
    for (int o = 16; o > 0; o >>= 1) v += __shfl_down(v, o, 32);
    return v;
}

// ---------------------------------------------------------------------------
// K0: constant envelopes (single block) + the k -> flattened-latz-offset
// table used by the WMMA B-gather: offt[k] = (k/55)*182 + k%55.
// ---------------------------------------------------------------------------
__global__ void k0_envelopes(float* __restrict__ rf_env, float* __restrict__ sre,
                             float* __restrict__ lri_env, int* __restrict__ offt) {
    __shared__ float red[1024];
    const int t = threadIdx.x;

    // --- gather-offset table (covers padded K; guard rows keep it in-bounds)
    for (int k = t; k < KPAD; k += 1024) {
        int dy = k / LRI_K;
        offt[k] = dy * ZDIM + (k - dy * LRI_K);
    }

    // --- rf envelope: gauss(15, std=3) * circle(r=7.5), L2-normalized ---
    float rfv = 0.f;
    if (t < RF_J) {
        float dr = (float)(t / RF_K) - 7.f, dc = (float)(t % RF_K) - 7.f;
        float d2 = dr*dr + dc*dc;
        rfv = __expf(-d2 / 18.f) * ((d2 < 56.25f) ? 1.f : 0.f);
    }
    red[t] = rfv * rfv;
    __syncthreads();
    #pragma unroll
    for (int o = 512; o > 0; o >>= 1) { if (t < o) red[t] += red[t + o]; __syncthreads(); }
    float rfinv = rsqrtf(red[0] + 1e-30f);
    if (t < RF_J) rf_env[t] = rfv * rfinv;
    __syncthreads();

    // --- sre: gauss(5, std=1) * circle(r=2.5), sum-normalized ---
    float sv = 0.f;
    if (t < SRE_KS*SRE_KS) {
        float dr = (float)(t / SRE_KS) - 2.f, dc = (float)(t % SRE_KS) - 2.f;
        float d2 = dr*dr + dc*dc;
        sv = __expf(-d2 * 0.5f) * ((d2 < 6.25f) ? 1.f : 0.f);
    }
    red[t] = sv;
    __syncthreads();
    #pragma unroll
    for (int o = 512; o > 0; o >>= 1) { if (t < o) red[t] += red[t + o]; __syncthreads(); }
    float sinv = 1.f / (red[0] + 1e-30f);
    if (t < SRE_KS*SRE_KS) sre[t] = sv * sinv;
    __syncthreads();

    // --- lri: gauss(55, std=9) * (1 - gauss(std=1)) * circle(r=27.5), /max ---
    float mx = 0.f;
    for (int k = t; k < LRI_J; k += 1024) {
        float dr = (float)(k / LRI_K) - 27.f, dc = (float)(k % LRI_K) - 27.f;
        float d2 = dr*dr + dc*dc;
        float g  = __expf(-d2 / 162.f);
        float m  = 1.f - __expf(-d2 * 0.5f);
        float cr = (d2 < 756.25f) ? 1.f : 0.f;
        float v  = g * m * cr;
        lri_env[k] = v;
        mx = fmaxf(mx, v);
    }
    red[t] = mx;
    __syncthreads();
    #pragma unroll
    for (int o = 512; o > 0; o >>= 1) { if (t < o) red[t] = fmaxf(red[t], red[t + o]); __syncthreads(); }
    float minv = 1.f / (red[0] + 1e-30f);
    for (int k = t; k < LRI_J; k += 1024) lri_env[k] *= minv;
}

// ---------------------------------------------------------------------------
// K1: affinity. One wave per unit; bilinear 128->170 resample fused with the
// 15x15 dilated unfold and the rf normalizations.
// ---------------------------------------------------------------------------
__global__ void k1_aff(const float* __restrict__ xin, const float* __restrict__ rfs,
                       const float* __restrict__ rf_env,
                       float* __restrict__ aff, float* __restrict__ lat) {
    const int wid = threadIdx.x >> 5, lane = threadIdx.x & 31;
    const int l = blockIdx.x * 8 + wid;
    const int r = l >> 7, c = l & 127;
    const float* w = rfs + (size_t)l * RF_J;

    float num = 0.f, den = 0.f, ssum = 0.f;
    const float sc = 128.f / 170.f;
    for (int j = lane; j < RF_J; j += 32) {
        int dy = j / RF_K, dx = j % RF_K;
        float fy = ((float)(r + dy*3) + 0.5f) * sc - 0.5f;
        float fx = ((float)(c + dx*3) + 0.5f) * sc - 0.5f;
        fy = fminf(fmaxf(fy, 0.f), 127.f);
        fx = fminf(fmaxf(fx, 0.f), 127.f);
        int y0 = (int)fy, x0 = (int)fx;
        int y1 = min(y0 + 1, 127), x1 = min(x0 + 1, 127);
        float wy = fy - (float)y0, wx = fx - (float)x0;
        float v00 = xin[y0*128 + x0], v01 = xin[y0*128 + x1];
        float v10 = xin[y1*128 + x0], v11 = xin[y1*128 + x1];
        float xv = (v00*(1.f-wx) + v01*wx)*(1.f-wy) + (v10*(1.f-wx) + v11*wx)*wy;
        float wj = w[j], e = rf_env[j];
        num += xv * wj * e;
        den += e * wj;
        ssum += wj;
    }
    num = wred(num); den = wred(den); ssum = wred(ssum);
    if (lane == 0) {
        float scale = ((float)RF_J * 0.1f) / (ssum + 1e-30f);   // renorm to sum = J*0.1
        float a = (num * scale) / (den * scale + 1e-7f);
        aff[l] = a;
        lat[l] = fmaxf(a, 0.f);                                  // lat0 = relu(aff)
    }
}

// ---------------------------------------------------------------------------
// K2: per-unit lateral-weight normalization coefficient.
// lat_w[l,k] = w[l,k]*lri[k]*coef[l],  coef = scale1 / (sum(w*scale1*lri)+1e-7)
// ---------------------------------------------------------------------------
__global__ void k2_latw_coef(const float* __restrict__ lw, const float* __restrict__ lri,
                             float* __restrict__ coef) {
    const int wid = threadIdx.x >> 5, lane = threadIdx.x & 31;
    const int l = blockIdx.x * 8 + wid;
    const float* w = lw + (size_t)l * LRI_J;
    float s1 = 0.f, s2 = 0.f;
    for (int k = lane; k < LRI_J; k += 32) { float wv = w[k]; s1 += wv; s2 += wv * lri[k]; }
    s1 = wred(s1); s2 = wred(s2);
    if (lane == 0) {
        float scale1 = ((float)LRI_J * 0.1f) / (s1 + 1e-30f);
        coef[l] = scale1 / (s2 * scale1 + 1e-7f);
    }
}

// ---------------------------------------------------------------------------
// K3: swizzle lateral weights into the exact 16x16x32 f16 A-matrix lane
// layout (ISA 7.12.2): lanes 0-15 hold M, K={0..7,16..23}; lanes 16-31 hold
// K={8..15,24..31}. One record = 16 halves per (group, chunk, lane).
// 99.5 MB total -> L2 resident across the 10 iterations.
// ---------------------------------------------------------------------------
__global__ void k3_swizzleA(const float* __restrict__ lw, const float* __restrict__ lri,
                            const float* __restrict__ coef, _Float16* __restrict__ A) {
    const int idx  = blockIdx.x * blockDim.x + threadIdx.x;  // (g, chunk, lane)
    const int lane = idx & 31;
    const int ck   = (idx >> 5) % KCHUNKS;
    const int g    = idx / (32 * KCHUNKS);
    const int m    = lane & 15;
    const size_t l = (size_t)g * 16 + m;
    const float* w = lw + l * LRI_J;
    const float cf = coef[l];
    _Float16* dst = A + (size_t)idx * 16;
    const bool lo = lane < 16;
    #pragma unroll
    for (int i = 0; i < 16; i++) {
        int Kloc = lo ? (i < 8 ? i : 8 + i) : (i < 8 ? 8 + i : 16 + i);
        int k = ck * 32 + Kloc;
        float v = 0.f;
        if (k < LRI_J) v = w[k] * lri[k] * cf;
        dst[i] = (_Float16)v;
    }
}

// ---------------------------------------------------------------------------
// K4: 5x5 SRE conv with reflect padding; writes latc (f32) and the 184x182
// zero-padded f16 latz (2 guard rows at the bottom keep the padded-K WMMA
// gathers in-bounds and reading 0.0, so the hot loop needs no predicates).
// ---------------------------------------------------------------------------
__global__ void k4_sre_conv(const float* __restrict__ lat, const float* __restrict__ sre,
                            float* __restrict__ latc, _Float16* __restrict__ latz) {
    const int p = blockIdx.x * blockDim.x + threadIdx.x;
    if (p >= ZROWS * ZDIM) return;
    const int zr = p / ZDIM, zc = p % ZDIM;
    _Float16 hv = (_Float16)0.f;
    if (zr >= ZPAD && zr < ZPAD + SHEET && zc >= ZPAD && zc < ZPAD + SHEET) {
        const int r = zr - ZPAD, c = zc - ZPAD;
        float s = 0.f;
        #pragma unroll
        for (int a = 0; a < SRE_KS; a++) {
            int rr = r - 2 + a;
            rr = rr < 0 ? -rr : (rr > 127 ? 254 - rr : rr);   // jnp reflect
            #pragma unroll
            for (int b = 0; b < SRE_KS; b++) {
                int cc = c - 2 + b;
                cc = cc < 0 ? -cc : (cc > 127 ? 254 - cc : cc);
                s += sre[a*SRE_KS + b] * lat[rr*128 + cc];
            }
        }
        latc[r*128 + c] = s;
        hv = (_Float16)s;
    }
    latz[p] = hv;
}

// ---------------------------------------------------------------------------
// K5: lateral inhibition via WMMA. One wave = 16 column-adjacent units.
// D = A(16x32 weights) x B(32x16 shifted windows) over 95 K-chunks;
// lat_neg = diag(D). The gather offsets are LDS-resident (4x ds_load_b128,
// broadcast per half-wave); each B element is one v_add + global_load_u16
// against the uniform latz base (saddr + scaled voffset) -> no 64-bit chains.
// ---------------------------------------------------------------------------
__global__ void __launch_bounds__(128)
k5_lat_wmma(const _Float16* __restrict__ A, const _Float16* __restrict__ latz,
            const int* __restrict__ offt, const float* __restrict__ latc,
            const float* __restrict__ aff, float* __restrict__ lat) {
    __shared__ int soff[KPAD];                 // 12160 B of 320 KB LDS
    for (int i = threadIdx.x; i < KPAD; i += 128) soff[i] = offt[i];
    __syncthreads();

    const int wid = threadIdx.x >> 5, lane = threadIdx.x & 31;
    const int g = blockIdx.x * 4 + wid;
    const int r = g >> 3;               // 8 groups per sheet row
    const int cbase = (g & 7) << 4;
    const int n  = lane & 15;           // B column = unit within group
    const int kb = (lane < 16) ? 0 : 8;
    const int base = r * ZDIM + cbase + n;     // lane's latz element offset

    const v16h* Ap = (const v16h*)(A + ((size_t)g * KCHUNKS * 32 + lane) * 16);

    v8f acc = {};
    for (int ck = 0; ck < KCHUNKS; ++ck) {
        v16h a = Ap[(size_t)ck * 32];
        if (ck + 1 < KCHUNKS) __builtin_prefetch(&Ap[(size_t)(ck + 1) * 32], 0, 3);

        // 16 gather offsets for this chunk: 32B-aligned LDS vector loads.
        const int kbase = ck * 32 + kb;        // multiple of 8
        const int4* ot = (const int4*)&soff[kbase];
        int4 o0 = ot[0], o1 = ot[1], o2 = ot[4], o3 = ot[5];
        int off[16] = { o0.x, o0.y, o0.z, o0.w,  o1.x, o1.y, o1.z, o1.w,
                        o2.x, o2.y, o2.z, o2.w,  o3.x, o3.y, o3.z, o3.w };

        // B[k, n] = latz[r + k/55, cbase + n + k%55] = latz[base + offt[k]]
        v16h b;
        #pragma unroll
        for (int e = 0; e < 16; e++) b[e] = latz[base + off[e]];

        acc = __builtin_amdgcn_wmma_f32_16x16x32_f16(
            /*neg_a=*/false, a, /*neg_b=*/false, b,
            /*c_mod=*/(short)0, acc, /*reuse_a=*/false, /*reuse_b=*/false);
    }

    // diag(D): lanes 0-7 hold M=N=lane in acc[lane]; lanes 24-31 hold
    // M=N=lane-16 in acc[lane-24]  (ISA 7.12.2 C/D layout).
    int m = -1;
    if (lane < 8)        m = lane;
    else if (lane >= 24) m = lane - 16;
    if (m >= 0) {
        int vi = (m < 8) ? m : m - 8;
        float d = acc[0];
        d = (vi == 1) ? acc[1] : d;  d = (vi == 2) ? acc[2] : d;
        d = (vi == 3) ? acc[3] : d;  d = (vi == 4) ? acc[4] : d;
        d = (vi == 5) ? acc[5] : d;  d = (vi == 6) ? acc[6] : d;
        d = (vi == 7) ? acc[7] : d;
        const int l = r * 128 + cbase + m;
        float v = (latc[l] - d) * 1.0f + aff[l] * 0.5f;   // STRENGTH, AFF_SCALER
        v = fmaxf(v, 0.f);
        lat[l] = tanhf(v);                                 // /SATURATION==1
    }
}

__global__ void k6_copy(const float* __restrict__ lat, float* __restrict__ out) {
    int i = blockIdx.x * blockDim.x + threadIdx.x;
    if (i < L_UNITS) out[i] = lat[i];
}

// ---------------------------------------------------------------------------
// Workspace layout (bytes). Requires ~100 MB of d_ws (f16 swizzled lateral
// weights dominate: 1024 groups * 95 chunks * 32 lanes * 16 halves * 2 B).
// ---------------------------------------------------------------------------
#define WS_RFENV   0u
#define WS_SRE     1024u
#define WS_LRI     2048u                       // 12100 B
#define WS_OFFT    14336u                      // 3040 ints = 12160 B
#define WS_COEF    32768u                      // 65536 B
#define WS_AFF     (WS_COEF + 65536u)
#define WS_LAT     (WS_AFF + 65536u)
#define WS_LATC    (WS_LAT + 65536u)
#define WS_LATZ    (WS_LATC + 65536u)          // 184*182 halves = 66976 B
#define WS_ASWZ    (WS_LATZ + 67584u)          // 32B-aligned

extern "C" void kernel_launch(void* const* d_in, const int* in_sizes, int n_in,
                              void* d_out, int out_size, void* d_ws, size_t ws_size,
                              hipStream_t stream) {
    (void)in_sizes; (void)n_in; (void)out_size; (void)ws_size;
    const float* xin  = (const float*)d_in[0];   // [1,1,128,128]
    const float* rfs  = (const float*)d_in[1];   // [16384,225,1]
    const float* latw = (const float*)d_in[2];   // [16384,3025,1]
    float* out = (float*)d_out;

    char* ws = (char*)d_ws;
    float*    rf_env = (float*)(ws + WS_RFENV);
    float*    sre    = (float*)(ws + WS_SRE);
    float*    lri    = (float*)(ws + WS_LRI);
    int*      offt   = (int*)(ws + WS_OFFT);
    float*    coef   = (float*)(ws + WS_COEF);
    float*    aff    = (float*)(ws + WS_AFF);
    float*    lat    = (float*)(ws + WS_LAT);
    float*    latc   = (float*)(ws + WS_LATC);
    _Float16* latz   = (_Float16*)(ws + WS_LATZ);
    _Float16* Aswz   = (_Float16*)(ws + WS_ASWZ);

    k0_envelopes<<<1, 1024, 0, stream>>>(rf_env, sre, lri, offt);
    k1_aff<<<L_UNITS / 8, 256, 0, stream>>>(xin, rfs, rf_env, aff, lat);
    k2_latw_coef<<<L_UNITS / 8, 256, 0, stream>>>(latw, lri, coef);
    {
        const int total = NGROUPS * KCHUNKS * 32;        // 3,112,960 records
        k3_swizzleA<<<total / 256, 256, 0, stream>>>(latw, lri, coef, Aswz);
    }
    for (int it = 0; it < LAT_ITERS; ++it) {
        k4_sre_conv<<<(ZROWS * ZDIM + 255) / 256, 256, 0, stream>>>(lat, sre, latc, latz);
        k5_lat_wmma<<<NGROUPS / 4, 128, 0, stream>>>(Aswz, latz, offt, latc, aff, lat);
    }
    k6_copy<<<L_UNITS / 256, 256, 0, stream>>>(lat, out);
}
